// EncoderRNN_90606630076879
// MI455X (gfx1250) — matrix operand order repaired and checked by reading, not compiled, not run on hardware
//
#include <hip/hip_runtime.h>
#include <math.h>

// ---------------------------------------------------------------------------
// CDNA5 (gfx1250) bf16-WMMA implementation of a 3-layer "bidirectional" GRU.
//   B=128, T=64, IN=256, H=1024, L=3, both directions run forward in time.
// Matmuls use V_WMMA_F32_16X16X32_BF16 (f32 accumulate); recurrent carry h
// kept in f32. Weights converted to bf16 each call (~91MB, resident in L2).
// ---------------------------------------------------------------------------

typedef __attribute__((ext_vector_type(16))) __bf16 v16bf;
typedef __attribute__((ext_vector_type(8)))  __bf16 v8bf;
typedef __attribute__((ext_vector_type(8)))  float  v8f;

#define GRU_B  128
#define GRU_T  64
#define GRU_IN 256
#define GRU_H  1024
#define GRU_H2 2048
#define GRU_H3 3072

// A-matrix 16x32 bf16 fragment (M row = lane&15). Per ISA table: lane half h:
// elements 0-7  <- K in [8h,   8h+8)
// elements 8-15 <- K in [16+8h,16+8h+8)
static __device__ __forceinline__ v16bf load_a_frag(const __bf16* __restrict__ base, int ld) {
  const int lane = threadIdx.x & 31;
  const int m = lane & 15, h = lane >> 4;
  const __bf16* p = base + (size_t)m * ld + 8 * h;
  v8bf lo = *(const v8bf*)(p);
  v8bf hi = *(const v8bf*)(p + 16);
  v16bf f;
#pragma unroll
  for (int i = 0; i < 8; ++i) { f[i] = lo[i]; f[i + 8] = hi[i]; }
  return f;
}

// B-matrix 32x16 bf16 fragment taken from row-major W[n][k] (NT GEMM: B[k][n]=W[n][k]).
// Lane = column n (lane&15); lane half h: elements 0-7 <- K in [8h,8h+8), 8-15 <- K in [16+8h,16+8h+8).
static __device__ __forceinline__ v16bf load_b_frag(const __bf16* __restrict__ wbase, int ld) {
  const int lane = threadIdx.x & 31;
  const int n = lane & 15, h = lane >> 4;
  const __bf16* p = wbase + (size_t)n * ld + 8 * h;
  v8bf lo = *(const v8bf*)(p);
  v8bf hi = *(const v8bf*)(p + 16);
  v16bf f;
#pragma unroll
  for (int i = 0; i < 8; ++i) { f[i] = lo[i]; f[i + 8] = hi[i]; }
  return f;
}

static __device__ __forceinline__ v8f wmma_bf16(v16bf a, v16bf b, v8f c) {
  return __builtin_amdgcn_wmma_f32_16x16x32_bf16(false, a, false, b, (short)0, c,
                                                 false, false);
}

static __device__ __forceinline__ float sigmoidf_dev(float x) {
  return 1.0f / (1.0f + __expf(-x));
}

// ---------------------------------------------------------------------------
// elementwise f32 -> bf16
__global__ void cvt_f32_bf16(const float* __restrict__ in, __bf16* __restrict__ out, int n) {
  int i = blockIdx.x * blockDim.x + threadIdx.x;
  if (i < n) out[i] = (__bf16)in[i];
}

// x: (B,T,IN) f32  ->  seq0: (T,B,IN) bf16
__global__ void transpose_x(const float* __restrict__ x, __bf16* __restrict__ seq0) {
  int i = blockIdx.x * blockDim.x + threadIdx.x;
  if (i >= GRU_T * GRU_B * GRU_IN) return;
  int d = i & (GRU_IN - 1);
  int tb = i >> 8;               // IN == 256
  int b = tb & (GRU_B - 1);
  int t = tb >> 7;               // B == 128
  seq0[i] = (__bf16)x[((size_t)b * GRU_T + t) * GRU_IN + d];
}

// ---------------------------------------------------------------------------
// gi = X (T*B, K) @ Wih[k] (3072, K)^T + b_ih[k]   ->  gi (T,2,B,3072) bf16
// grid: (T*B/16, 3072/256, 2), block = 128 (4 waves, each wave: 16x64 tile)
__global__ void gi_gemm(const __bf16* __restrict__ X, int ldx,
                        const __bf16* __restrict__ Wih,   // (2,3072,ldx)
                        const float*  __restrict__ bih,   // (2,3072)
                        __bf16* __restrict__ gi) {
  const int wave = threadIdx.x >> 5;
  const int k = blockIdx.z;
  const int m0 = blockIdx.x * 16;
  const int n0 = (blockIdx.y * 4 + wave) * 64;
  const __bf16* W = Wih + (size_t)k * GRU_H3 * ldx;
  const __bf16* A = X + (size_t)m0 * ldx;

  v8f acc[4] = {v8f{}, v8f{}, v8f{}, v8f{}};
  for (int kk = 0; kk < ldx; kk += 32) {
    v16bf a = load_a_frag(A + kk, ldx);
#pragma unroll
    for (int st = 0; st < 4; ++st) {
      v16bf b = load_b_frag(W + (size_t)(n0 + st * 16) * ldx + kk, ldx);
      acc[st] = wmma_bf16(a, b, acc[st]);
    }
  }

  const int lane = threadIdx.x & 31;
  const int nlane = lane & 15, mh = lane >> 4;
#pragma unroll
  for (int st = 0; st < 4; ++st) {
    const int n = n0 + st * 16 + nlane;
    const float bias = bih[k * GRU_H3 + n];
#pragma unroll
    for (int v = 0; v < 8; ++v) {
      const int m = m0 + v + 8 * mh;
      const int b_ = m & (GRU_B - 1);
      const int t = m >> 7;
      gi[((size_t)(t * 2 + k) * GRU_B + b_) * GRU_H3 + n] = (__bf16)(acc[st][v] + bias);
    }
  }
}

// ---------------------------------------------------------------------------
// One GRU time step: gh = h @ Whh^T (+b_hh), fused gates, write h_new.
// grid: (H/16, B/16, 2 dirs), block = 32 (1 wave). Each wave: 16 batch rows x
// 16 hidden cols, three gate accumulators (r,z,n) sharing the A fragment.
__global__ void gru_step(const __bf16* __restrict__ hbf_in,   // (2,B,H) bf16
                         const float*  __restrict__ hf_in,    // (2,B,H) f32
                         const __bf16* __restrict__ Whh,      // (2,3H,H) bf16
                         const float*  __restrict__ bhh,      // (2,3H)
                         const __bf16* __restrict__ gi,       // (T,2,B,3H) bf16
                         __bf16* __restrict__ hbf_out,
                         float*  __restrict__ hf_out,
                         __bf16* __restrict__ seq_out,        // (T,B,2H) bf16
                         const int* __restrict__ seq_len,
                         float* __restrict__ out_hidden,      // (B,2H) f32
                         int t, int is_last) {
  const int j0 = blockIdx.x * 16;
  const int b0 = blockIdx.y * 16;
  const int k = blockIdx.z;

  const __bf16* A  = hbf_in + ((size_t)k * GRU_B + b0) * GRU_H;
  const __bf16* Wr = Whh + ((size_t)k * GRU_H3 + 0 * GRU_H + j0) * GRU_H;
  const __bf16* Wz = Whh + ((size_t)k * GRU_H3 + 1 * GRU_H + j0) * GRU_H;
  const __bf16* Wn = Whh + ((size_t)k * GRU_H3 + 2 * GRU_H + j0) * GRU_H;

  v8f ar = {}, az = {}, an = {};
  for (int kk = 0; kk < GRU_H; kk += 32) {
    v16bf a = load_a_frag(A + kk, GRU_H);
    ar = wmma_bf16(a, load_b_frag(Wr + kk, GRU_H), ar);
    az = wmma_bf16(a, load_b_frag(Wz + kk, GRU_H), az);
    an = wmma_bf16(a, load_b_frag(Wn + kk, GRU_H), an);
  }

  const int lane = threadIdx.x & 31;
  const int j = j0 + (lane & 15);
  const int mh = lane >> 4;
  const float br = bhh[k * GRU_H3 + j];
  const float bz = bhh[k * GRU_H3 + GRU_H + j];
  const float bn = bhh[k * GRU_H3 + 2 * GRU_H + j];
  const __bf16* gib = gi + (size_t)(t * 2 + k) * GRU_B * GRU_H3;

#pragma unroll
  for (int v = 0; v < 8; ++v) {
    const int b = b0 + v + 8 * mh;
    const float gr = (float)gib[(size_t)b * GRU_H3 + j];
    const float gz = (float)gib[(size_t)b * GRU_H3 + GRU_H + j];
    const float gn = (float)gib[(size_t)b * GRU_H3 + 2 * GRU_H + j];
    const float r = sigmoidf_dev(gr + ar[v] + br);
    const float z = sigmoidf_dev(gz + az[v] + bz);
    const float n = tanhf(gn + r * (an[v] + bn));
    const size_t hidx = ((size_t)k * GRU_B + b) * GRU_H + j;
    const float hold = hf_in[hidx];
    const float hnew = (1.0f - z) * n + z * hold;
    hf_out[hidx] = hnew;
    hbf_out[hidx] = (__bf16)hnew;
    seq_out[((size_t)t * GRU_B + b) * GRU_H2 + k * GRU_H + j] = (__bf16)hnew;
    if (is_last && t == seq_len[b] - 1)
      out_hidden[(size_t)b * GRU_H2 + k * GRU_H + j] = hnew;
  }
}

// ---------------------------------------------------------------------------
extern "C" void kernel_launch(void* const* d_in, const int* in_sizes, int n_in,
                              void* d_out, int out_size, void* d_ws, size_t ws_size,
                              hipStream_t stream) {
  (void)in_sizes; (void)n_in; (void)out_size; (void)ws_size;
  // dict order: input_tensor, seq_len, w_ih[0..2], w_hh[0..2], b_ih[0..2], b_hh[0..2]
  const float* x = (const float*)d_in[0];
  const int* seq_len = (const int*)d_in[1];
  const float* wih[3] = {(const float*)d_in[2], (const float*)d_in[3], (const float*)d_in[4]};
  const float* whh[3] = {(const float*)d_in[5], (const float*)d_in[6], (const float*)d_in[7]};
  const float* bih[3] = {(const float*)d_in[8], (const float*)d_in[9], (const float*)d_in[10]};
  const float* bhh[3] = {(const float*)d_in[11], (const float*)d_in[12], (const float*)d_in[13]};

  const int din[3] = {GRU_IN, GRU_H2, GRU_H2};

  char* p = (char*)d_ws;
  auto alloc = [&](size_t bytes) -> char* {
    char* r = p;
    p += (bytes + 255) & ~(size_t)255;
    return r;
  };

  __bf16* wih_b[3];
  __bf16* whh_b[3];
  for (int l = 0; l < 3; ++l) wih_b[l] = (__bf16*)alloc((size_t)2 * GRU_H3 * din[l] * 2);
  for (int l = 0; l < 3; ++l) whh_b[l] = (__bf16*)alloc((size_t)2 * GRU_H3 * GRU_H * 2);
  __bf16* seqA = (__bf16*)alloc((size_t)GRU_T * GRU_B * GRU_H2 * 2);
  __bf16* seqB = (__bf16*)alloc((size_t)GRU_T * GRU_B * GRU_H2 * 2);
  __bf16* gi   = (__bf16*)alloc((size_t)GRU_T * 2 * GRU_B * GRU_H3 * 2);
  float*  hf[2];
  __bf16* hb[2];
  for (int i = 0; i < 2; ++i) {
    hf[i] = (float*)alloc((size_t)2 * GRU_B * GRU_H * 4);
    hb[i] = (__bf16*)alloc((size_t)2 * GRU_B * GRU_H * 2);
  }

  // weight + input conversion (every call; no cross-call state allowed)
  for (int l = 0; l < 3; ++l) {
    const int n1 = 2 * GRU_H3 * din[l];
    cvt_f32_bf16<<<(n1 + 255) / 256, 256, 0, stream>>>(wih[l], wih_b[l], n1);
    const int n2 = 2 * GRU_H3 * GRU_H;
    cvt_f32_bf16<<<(n2 + 255) / 256, 256, 0, stream>>>(whh[l], whh_b[l], n2);
  }
  transpose_x<<<(GRU_T * GRU_B * GRU_IN + 255) / 256, 256, 0, stream>>>(x, seqA);

  __bf16* in_seq = seqA;
  __bf16* out_seq = seqB;
  for (int l = 0; l < 3; ++l) {
    // parallel input projection for all timesteps, both directions
    gi_gemm<<<dim3(GRU_T * GRU_B / 16, GRU_H3 / 256, 2), 128, 0, stream>>>(
        in_seq, din[l], wih_b[l], bih[l], gi);

    hipMemsetAsync(hf[0], 0, (size_t)2 * GRU_B * GRU_H * 4, stream);
    hipMemsetAsync(hb[0], 0, (size_t)2 * GRU_B * GRU_H * 2, stream);

    for (int t = 0; t < GRU_T; ++t) {
      const int a = t & 1, o = (t & 1) ^ 1;
      gru_step<<<dim3(GRU_H / 16, GRU_B / 16, 2), 32, 0, stream>>>(
          hb[a], hf[a], whh_b[l], bhh[l], gi, hb[o], hf[o], out_seq, seq_len,
          (float*)d_out, t, (l == 2) ? 1 : 0);
    }
    __bf16* tmp = in_seq; in_seq = out_seq; out_seq = tmp;
  }
}
